// SAHead_16965120819424
// MI455X (gfx1250) — matrix operand order repaired and checked by reading, not compiled, not run on hardware
//
#include <hip/hip_runtime.h>

typedef __attribute__((ext_vector_type(16))) __bf16        v16bf;
typedef __attribute__((ext_vector_type(8)))  float         v8f;
typedef __attribute__((ext_vector_type(4)))  int           v4i;
typedef __attribute__((ext_vector_type(8)))  unsigned int  v8ui;

#define BB 4
#define CC 256
#define C8 32
#define HH 128
#define WW 128
#define PP (HH*WW)        // 16384
#define HS 64
#define NS (HS*HS)        // 4096
#define EPS 1e-5f

#if __has_builtin(__builtin_amdgcn_global_load_async_to_lds_b128) && \
    __has_builtin(__builtin_amdgcn_s_wait_asynccnt)
#define HAVE_ASYNC_LDS 1
#else
#define HAVE_ASYNC_LDS 0
#endif

// round-half-up f32 -> bf16 bits (2 VALU)
static __device__ __forceinline__ unsigned short f2bfbits(float f) {
    return (unsigned short)((__builtin_bit_cast(unsigned int, f) + 0x8000u) >> 16);
}
// pack two f32 -> dword {hi.bf16, lo.bf16}: 2x add + 1x v_perm_b32
static __device__ __forceinline__ unsigned int pack_bf16(float lo, float hi) {
    const unsigned int a = __builtin_bit_cast(unsigned int, lo) + 0x8000u;
    const unsigned int b = __builtin_bit_cast(unsigned int, hi) + 0x8000u;
    return __builtin_amdgcn_perm(b, a, 0x07060302u);   // {b[31:16], a[31:16]}
}

#if HAVE_ASYNC_LDS
typedef __attribute__((address_space(1))) v4i* as1_v4i_ptr;   // global int4*
typedef __attribute__((address_space(3))) v4i* as3_v4i_ptr;   // LDS int4*
// 16B HBM -> LDS copy through the async path (ASYNCcnt-tracked).
static __device__ __forceinline__ void async_copy_b128(const void* g, void* l) {
    __builtin_amdgcn_global_load_async_to_lds_b128(
        (as1_v4i_ptr)(unsigned long long)g,
        (as3_v4i_ptr)(unsigned int)(unsigned long long)l,
        0, 0);
}
static __device__ __forceinline__ void async_wait0() {
    __builtin_amdgcn_s_wait_asynccnt(0);
}
#else
static __device__ __forceinline__ void async_copy_b128(const void* g, void* l) {
    *(v4i*)l = *(const v4i*)g;
}
static __device__ __forceinline__ void async_wait0() {}
#endif

// --- 16-lane reductions via DPP16 (no LDS, no waits) -----------------------
static __device__ __forceinline__ float dpp_bcast(float x, int ctrl_tag) {
    int v = __builtin_bit_cast(int, x), t;
    switch (ctrl_tag) {
    case 0: t = __builtin_amdgcn_update_dpp(0, v, 0xB1,  0xF, 0xF, true); break; // quad_perm xor1
    case 1: t = __builtin_amdgcn_update_dpp(0, v, 0x4E,  0xF, 0xF, true); break; // quad_perm xor2
    case 2: t = __builtin_amdgcn_update_dpp(0, v, 0x141, 0xF, 0xF, true); break; // row_half_mirror
    default:t = __builtin_amdgcn_update_dpp(0, v, 0x140, 0xF, 0xF, true); break; // row_mirror
    }
    return __builtin_bit_cast(float, t);
}
static __device__ __forceinline__ float red16_max(float x) {
    x = fmaxf(x, dpp_bcast(x, 0)); x = fmaxf(x, dpp_bcast(x, 1));
    x = fmaxf(x, dpp_bcast(x, 2)); x = fmaxf(x, dpp_bcast(x, 3));
    return x;
}
static __device__ __forceinline__ float red16_sum(float x) {
    x += dpp_bcast(x, 0); x += dpp_bcast(x, 1);
    x += dpp_bcast(x, 2); x += dpp_bcast(x, 3);
    return x;
}

// ---------------------------------------------------------------------------
// conv1x1 GEMM: acc[b,o,p] = sum_c w[o,c]*in[b,c,p] (+ bias[o])
// OUT_MODE: 0 = f32 (B,O,P);  1 = bf16 (B,O,P);  2 = bf16 transposed (B,P,O)
// Block = NW waves covering NW adjacent o-tiles, sharing an LDS-staged 32x16
// activation tile per K-step (async HBM->LDS).
// grid = (P/16, O/(16*NW), B), block = 32*NW
// ---------------------------------------------------------------------------
template<int OUT_MODE, bool HAS_BIAS>
__global__ void gemm_conv1x1(const float* __restrict__ in, const float* __restrict__ w,
                             const float* __restrict__ bias, void* __restrict__ out,
                             int Cin, int O, int P) {
    __shared__ float btile[32][20];            // [k][p], padded row (mult of 4 floats)

    const int tid  = threadIdx.x;
    const int nw   = blockDim.x >> 5;
    const int wave = tid >> 5, lane = tid & 31;
    const int half = lane >> 4, lo = lane & 15;
    const int p0 = blockIdx.x * 16;
    const int o0 = (blockIdx.y * nw + wave) * 16;
    const int b  = blockIdx.z;
    const float* inb  = in + (size_t)b * Cin * P;
    const float* wrow = w + (size_t)(o0 + lo) * Cin;

    v8f acc;
#pragma unroll
    for (int i = 0; i < 8; ++i) acc[i] = 0.0f;

    for (int k0 = 0; k0 < Cin; k0 += 32) {
        __syncthreads();                       // protect btile reuse
        for (int t = tid; t < 128; t += blockDim.x) {
            const int kk = t >> 2, pc = (t & 3) * 4;
            async_copy_b128(inb + (size_t)(k0 + kk) * P + (p0 + pc), &btile[kk][pc]);
        }
        async_wait0();
        __syncthreads();

        if (k0 + 32 < Cin) __builtin_prefetch(wrow + k0 + 32, 0, 1);

        // A fragment: row o0+lo, two contiguous 8-float runs -> 4x b128 loads
        const float4* wp  = (const float4*)(wrow + k0 + 8 * half);
        const float4* wp2 = (const float4*)(wrow + k0 + 16 + 8 * half);
        const float4 a0 = wp[0],  a1 = wp[1];
        const float4 a2 = wp2[0], a3 = wp2[1];
        v8ui aw, bw;
        aw[0] = pack_bf16(a0.x, a0.y); aw[1] = pack_bf16(a0.z, a0.w);
        aw[2] = pack_bf16(a1.x, a1.y); aw[3] = pack_bf16(a1.z, a1.w);
        aw[4] = pack_bf16(a2.x, a2.y); aw[5] = pack_bf16(a2.z, a2.w);
        aw[6] = pack_bf16(a3.x, a3.y); aw[7] = pack_bf16(a3.z, a3.w);
#pragma unroll
        for (int g = 0; g < 8; ++g)            // B 32x16 layout: K = 2g(+1) + (g>=4?8:0) + 8*half
            bw[g] = pack_bf16(btile[2 * g + ((g >= 4) ? 8 : 0) + 8 * half][lo], btile[2 * g + 1 + ((g >= 4) ? 8 : 0) + 8 * half][lo]);
        const v16bf afr = __builtin_bit_cast(v16bf, aw);
        const v16bf bfr = __builtin_bit_cast(v16bf, bw);
        acc = __builtin_amdgcn_wmma_f32_16x16x32_bf16(false, afr, false, bfr,
                                                      (short)0, acc, false, false);
    }

#pragma unroll
    for (int r = 0; r < 8; ++r)
        if (HAS_BIAS) acc[r] += bias[o0 + r + 8 * half];

    if (OUT_MODE == 0) {
        float* outb = (float*)out + (size_t)b * O * P;
#pragma unroll
        for (int r = 0; r < 8; ++r)
            outb[(size_t)(o0 + r + 8 * half) * P + (p0 + lo)] = acc[r];
    } else if (OUT_MODE == 1) {                // bf16, natural (B,O,P)
        unsigned short* outb = (unsigned short*)out + (size_t)b * O * P;
#pragma unroll
        for (int r = 0; r < 8; ++r)
            outb[(size_t)(o0 + r + 8 * half) * P + (p0 + lo)] = f2bfbits(acc[r]);
    } else {                                   // bf16, transposed (B,P,O): one 16B store
        unsigned short* outb = (unsigned short*)out + (size_t)b * P * O;
        v4i st;
#pragma unroll
        for (int g = 0; g < 4; ++g)
            st[g] = (int)pack_bf16(acc[2 * g], acc[2 * g + 1]);
        *(v4i*)(outb + (size_t)(p0 + lo) * O + o0 + 8 * half) = st;
    }
}

// ---------------------------------------------------------------------------
// Per-channel mean / inv-std over (B, P). grid = C, block = 256.
// ---------------------------------------------------------------------------
__global__ void channel_stats(const float* __restrict__ x, float* __restrict__ mean,
                              float* __restrict__ istd, int P) {
    const int c = blockIdx.x, tid = threadIdx.x;
    float s = 0.f, ss = 0.f;
    for (int b = 0; b < BB; ++b) {
        const float* p = x + ((size_t)b * CC + c) * P;
        for (int i = tid; i < P; i += 256) { float v = p[i]; s += v; ss += v * v; }
    }
    __shared__ float sh0[256], sh1[256];
    sh0[tid] = s; sh1[tid] = ss; __syncthreads();
    for (int o = 128; o > 0; o >>= 1) {
        if (tid < o) { sh0[tid] += sh0[tid + o]; sh1[tid] += sh1[tid + o]; }
        __syncthreads();
    }
    if (tid == 0) {
        const float n = (float)BB * (float)P;
        const float m = sh0[0] / n;
        const float var = sh1[0] / n - m * m;
        mean[c] = m;
        istd[c] = rsqrtf(var + EPS);
    }
}

// ---------------------------------------------------------------------------
// BN + ReLU + 2x2 average pool: (B,C,128,128) -> (B,C,64,64)
// ---------------------------------------------------------------------------
__global__ void bn_relu_pool(const float* __restrict__ in, const float* __restrict__ g,
                             const float* __restrict__ bta, const float* __restrict__ mean,
                             const float* __restrict__ istd, float* __restrict__ out) {
    const int idx = blockIdx.x * blockDim.x + threadIdx.x;   // B*C*64*64
    const int x = idx & 63, y = (idx >> 6) & 63;
    const int c = (idx >> 12) & 255, b = idx >> 20;
    const float m = mean[c], is = istd[c], gg = g[c], bb = bta[c];
    const float* base = in + (((size_t)b * CC + c) << 14);
    float acc = 0.f;
#pragma unroll
    for (int dy = 0; dy < 2; ++dy)
#pragma unroll
        for (int dx = 0; dx < 2; ++dx) {
            float v = base[(2 * y + dy) * WW + (2 * x + dx)];
            v = (v - m) * is * gg + bb;
            acc += fmaxf(v, 0.f);
        }
    out[idx] = 0.25f * acc;
}

// ---------------------------------------------------------------------------
// Flash attention, N=4096. qT,kT:(B,N,32) bf16 transposed; v:(B,C,N) bf16.
// 8 waves/block: wave = (i-tile wi 0..3, channel-half ch 0..1); 8 acc tiles
// per wave (64 VGPRs, no spills). K/V chunks async-staged in LDS as bf16 ->
// all fragments are pure dword LDS loads, zero conversion in the inner loop.
// out = gamma * softmax(q^T k) @ v + hres (f32).   grid = (64, B), block 256.
// ---------------------------------------------------------------------------
__global__ void attention_flash(const unsigned short* __restrict__ qT,
                                const unsigned short* __restrict__ kT,
                                const unsigned short* __restrict__ vbf,
                                const float* __restrict__ hres,
                                const float* __restrict__ gamma_p, float* __restrict__ out) {
    __shared__ unsigned short kbufS[32][40];    // [j_local][d] bf16, 80B rows (16B mult)
    __shared__ unsigned short vbufS[256][40];   // [c][j_local] bf16
    __shared__ unsigned short pbuf[4][16][36];  // per-i-tile P tile [i][j] bf16 bits

    const int tid = threadIdx.x;
    const int wave = tid >> 5, lane = tid & 31;
    const int wi = wave & 3, ch = wave >> 2;    // i-tile, channel half
    const int half = lane >> 4, lo = lane & 15;
    const int b = blockIdx.y;
    const int i0 = blockIdx.x * 64 + wi * 16;

    const unsigned short* qTb = qT + (size_t)b * NS * C8;
    const unsigned short* kTb = kT + (size_t)b * NS * C8;
    const unsigned short* vbb = vbf + (size_t)b * CC * NS;

    // Q^T A-fragment: row i0+lo of qT, two 16B runs (d = 8h..8h+8, 16+8h..+8)
    v8ui qw;
    {
        const v4i q0 = *(const v4i*)(qTb + (size_t)(i0 + lo) * C8 + 8 * half);
        const v4i q1 = *(const v4i*)(qTb + (size_t)(i0 + lo) * C8 + 16 + 8 * half);
#pragma unroll
        for (int g = 0; g < 4; ++g) { qw[g] = (unsigned int)q0[g]; qw[g + 4] = (unsigned int)q1[g]; }
    }
    const v16bf qa = __builtin_bit_cast(v16bf, qw);

    v8f acc[8];
#pragma unroll
    for (int t = 0; t < 8; ++t)
#pragma unroll
        for (int r = 0; r < 8; ++r) acc[t][r] = 0.0f;

    float mrun[8], lrun[8];
#pragma unroll
    for (int r = 0; r < 8; ++r) { mrun[r] = -3.0e38f; lrun[r] = 0.0f; }

    for (int j0 = 0; j0 < NS; j0 += 32) {
        __syncthreads();   // protect kbufS/vbufS reuse
        // kT chunk: 32 rows x 64B = 128 b128 chunks; v chunk: 256 rows x 64B = 1024
        for (int t = tid; t < 128; t += 256) {
            const int j = t >> 2, dc = (t & 3) * 8;
            async_copy_b128(kTb + (size_t)(j0 + j) * C8 + dc, &kbufS[j][dc]);
        }
        for (int t = tid; t < 1024; t += 256) {
            const int c = t >> 2, jc = (t & 3) * 8;
            async_copy_b128(vbb + (size_t)c * NS + j0 + jc, &vbufS[c][jc]);
        }
        async_wait0();
        __syncthreads();

        // Energy B-fragments straight from LDS dwords (d pairs are adjacent)
        v8ui b0w, b1w;
#pragma unroll
        for (int g = 0; g < 8; ++g) {
            b0w[g] = *(const unsigned int*)&kbufS[lo]     [2 * g + ((g >= 4) ? 8 : 0) + 8 * half];
            b1w[g] = *(const unsigned int*)&kbufS[lo + 16][2 * g + ((g >= 4) ? 8 : 0) + 8 * half];
        }
        const v16bf b0 = __builtin_bit_cast(v16bf, b0w);
        const v16bf b1 = __builtin_bit_cast(v16bf, b1w);
        v8f z;
#pragma unroll
        for (int i = 0; i < 8; ++i) z[i] = 0.0f;
        v8f S0 = __builtin_amdgcn_wmma_f32_16x16x32_bf16(false, qa, false, b0, (short)0, z, false, false);
        v8f S1 = __builtin_amdgcn_wmma_f32_16x16x32_bf16(false, qa, false, b1, (short)0, z, false, false);

        // Online softmax: per row r (row i = r + 8*half); DPP16 row reductions.
#pragma unroll
        for (int r = 0; r < 8; ++r) {
            const float s0 = S0[r], s1 = S1[r];
            const float mx = red16_max(fmaxf(s0, s1));
            const float mnew = fmaxf(mrun[r], mx);
            const float corr = __expf(mrun[r] - mnew);
            const float p0 = __expf(s0 - mnew);
            const float p1 = __expf(s1 - mnew);
            const float rs = red16_sum(p0 + p1);
            lrun[r] = lrun[r] * corr + rs;
            mrun[r] = mnew;
#pragma unroll
            for (int t = 0; t < 8; ++t) acc[t][r] *= corr;
            if (ch == 0) {   // one channel-half publishes the shared P tile
                pbuf[wi][r + 8 * half][lo]      = f2bfbits(p0);
                pbuf[wi][r + 8 * half][lo + 16] = f2bfbits(p1);
            }
        }
        __syncthreads();   // pbuf visibility across the two channel halves

        // P A-fragment: adjacent bf16 pairs -> direct dword loads
        v8ui pw;
#pragma unroll
        for (int g = 0; g < 8; ++g) {
            const int kj = 2 * g + 16 * half;
            pw[g] = *(const unsigned int*)&pbuf[wi][lo][kj];
        }
        const v16bf pa = __builtin_bit_cast(v16bf, pw);
        // O += P (16x32) x V^T (32 j x 16 c): this wave's 8 channel tiles
#pragma unroll
        for (int t = 0; t < 8; ++t) {
            const int ct = (ch * 8 + t) * 16;
            v8ui vw;
#pragma unroll
            for (int g = 0; g < 8; ++g)
                vw[g] = *(const unsigned int*)&vbufS[ct + lo][2 * g + 16 * half];
            const v16bf vf = __builtin_bit_cast(v16bf, vw);
            acc[t] = __builtin_amdgcn_wmma_f32_16x16x32_bf16(false, pa, false, vf,
                                                             (short)0, acc[t], false, false);
        }
    }

    const float gamma = gamma_p[0];
#pragma unroll
    for (int t = 0; t < 8; ++t)
#pragma unroll
        for (int r = 0; r < 8; ++r) {
            const int c = (ch * 8 + t) * 16 + lo;
            const int i = i0 + r + 8 * half;
            const size_t o = ((size_t)b * CC + c) * NS + i;
            out[o] = gamma * (acc[t][r] / lrun[r]) + hres[o];
        }
}

// ---------------------------------------------------------------------------
// Elementwise BN + ReLU on (B,C,4096)
// ---------------------------------------------------------------------------
__global__ void bn_relu(const float* __restrict__ in, const float* __restrict__ g,
                        const float* __restrict__ bta, const float* __restrict__ mean,
                        const float* __restrict__ istd, float* __restrict__ out) {
    const int idx = blockIdx.x * blockDim.x + threadIdx.x;   // B*C*4096
    const int c = (idx >> 12) & 255;
    float v = (in[idx] - mean[c]) * istd[c] * g[c] + bta[c];
    out[idx] = fmaxf(v, 0.f);
}

// ---------------------------------------------------------------------------
// Bilinear 2x upsample, half-pixel centers, clamped taps: (B,C,64,64)->(B,C,128,128)
// ---------------------------------------------------------------------------
__global__ void upsample2x(const float* __restrict__ in, float* __restrict__ out) {
    const int idx = blockIdx.x * blockDim.x + threadIdx.x;   // B*C*128*128
    const int x = idx & 127, y = (idx >> 7) & 127;
    const int c = (idx >> 14) & 255, b = idx >> 22;
    const float sx = 0.5f * x - 0.25f, sy = 0.5f * y - 0.25f;
    const float fx0 = floorf(sx), fy0 = floorf(sy);
    const float wx = sx - fx0, wy = sy - fy0;
    const int x0 = max(0, (int)fx0), x1 = min(HS - 1, (int)fx0 + 1);
    const int y0 = max(0, (int)fy0), y1 = min(HS - 1, (int)fy0 + 1);
    const float* base = in + (((size_t)b * CC + c) << 12);
    const float v00 = base[y0 * HS + x0], v01 = base[y0 * HS + x1];
    const float v10 = base[y1 * HS + x0], v11 = base[y1 * HS + x1];
    const float top = v00 + wx * (v01 - v00);
    const float bot = v10 + wx * (v11 - v10);
    out[idx] = top + wy * (bot - top);
}

// ---------------------------------------------------------------------------
// Final: relu(BN3(y3) + x)  on (B,C,16384)
// ---------------------------------------------------------------------------
__global__ void bn_add_relu(const float* __restrict__ y3, const float* __restrict__ x,
                            const float* __restrict__ g, const float* __restrict__ bta,
                            const float* __restrict__ mean, const float* __restrict__ istd,
                            float* __restrict__ out) {
    const int idx = blockIdx.x * blockDim.x + threadIdx.x;   // B*C*16384
    const int c = (idx >> 14) & 255;
    float v = (y3[idx] - mean[c]) * istd[c] * g[c] + bta[c] + x[idx];
    out[idx] = fmaxf(v, 0.f);
}

// ---------------------------------------------------------------------------
extern "C" void kernel_launch(void* const* d_in, const int* in_sizes, int n_in,
                              void* d_out, int out_size, void* d_ws, size_t ws_size,
                              hipStream_t stream) {
    const float* x     = (const float*)d_in[0];
    const float* w1    = (const float*)d_in[1];
    const float* g1    = (const float*)d_in[2];
    const float* b1    = (const float*)d_in[3];
    const float* wq    = (const float*)d_in[4];
    const float* bq    = (const float*)d_in[5];
    const float* wk    = (const float*)d_in[6];
    const float* bk    = (const float*)d_in[7];
    const float* wv    = (const float*)d_in[8];
    const float* bv    = (const float*)d_in[9];
    const float* gamma = (const float*)d_in[10];
    const float* g2    = (const float*)d_in[11];
    const float* b2    = (const float*)d_in[12];
    const float* w3    = (const float*)d_in[13];
    const float* g3    = (const float*)d_in[14];
    const float* b3    = (const float*)d_in[15];
    float* out = (float*)d_out;

    const size_t SZ_IMG   = (size_t)BB * CC * PP;   // 16,777,216 f32
    const size_t SZ_SMALL = (size_t)BB * CC * NS;   //  4,194,304 f32
    const size_t SZ_QK16  = (size_t)BB * NS * C8;   //    524,288 bf16
    const size_t SZ_V16   = (size_t)BB * CC * NS;   //  4,194,304 bf16

    char* base = (char*)d_ws;
    float* bufA            = (float*)base;                    base += SZ_IMG * 4;   // y1 / upsampled
    float* hsmall          = (float*)base;                    base += SZ_SMALL * 4;
    float* h2              = (float*)base;                    base += SZ_SMALL * 4;
    float* h3              = (float*)base;                    base += SZ_SMALL * 4;
    float* y3              = (float*)base;                    base += SZ_IMG * 4;
    unsigned short* qTb    = (unsigned short*)base;           base += SZ_QK16 * 2;
    unsigned short* kTb    = (unsigned short*)base;           base += SZ_QK16 * 2;
    unsigned short* vbfb   = (unsigned short*)base;           base += SZ_V16 * 2;
    float* stats           = (float*)base;
    float* mean1 = stats,        *istd1 = stats + 256;
    float* mean2 = stats + 512,  *istd2 = stats + 768;
    float* mean3 = stats + 1024, *istd3 = stats + 1280;

    // 1. y1 = conv1x1(x, w1)
    gemm_conv1x1<0, false><<<dim3(PP / 16, CC / 64, BB), dim3(128), 0, stream>>>(
        x, w1, nullptr, bufA, CC, CC, PP);
    // 2. BN1 stats
    channel_stats<<<dim3(CC), dim3(256), 0, stream>>>(bufA, mean1, istd1, PP);
    // 3. h_small = avgpool2(relu(bn1(y1)))
    bn_relu_pool<<<dim3((unsigned)(SZ_SMALL / 256)), dim3(256), 0, stream>>>(
        bufA, g1, b1, mean1, istd1, hsmall);
    // 4. q,k,v projections -> fragment-friendly bf16 layouts
    gemm_conv1x1<2, true><<<dim3(NS / 16, 1, BB), dim3(64), 0, stream>>>(
        hsmall, wq, bq, qTb, CC, C8, NS);                      // qT (B,N,32) bf16
    gemm_conv1x1<2, true><<<dim3(NS / 16, 1, BB), dim3(64), 0, stream>>>(
        hsmall, wk, bk, kTb, CC, C8, NS);                      // kT (B,N,32) bf16
    gemm_conv1x1<1, true><<<dim3(NS / 16, CC / 64, BB), dim3(128), 0, stream>>>(
        hsmall, wv, bv, vbfb, CC, CC, NS);                     // v  (B,C,N) bf16
    // 5. h2 = gamma * attn(q,k,v) + h_small   (flash softmax)
    attention_flash<<<dim3(NS / 64, BB), dim3(256), 0, stream>>>(
        qTb, kTb, vbfb, hsmall, gamma, h2);
    // 6. BN2 + ReLU
    channel_stats<<<dim3(CC), dim3(256), 0, stream>>>(h2, mean2, istd2, NS);
    bn_relu<<<dim3((unsigned)(SZ_SMALL / 256)), dim3(256), 0, stream>>>(
        h2, g2, b2, mean2, istd2, h3);
    // 7. upsample 2x into bufA (y1 dead)
    upsample2x<<<dim3((unsigned)(SZ_IMG / 256)), dim3(256), 0, stream>>>(h3, bufA);
    // 8. y3 = conv1x1(h_up, w3)
    gemm_conv1x1<0, false><<<dim3(PP / 16, CC / 64, BB), dim3(128), 0, stream>>>(
        bufA, w3, nullptr, y3, CC, CC, PP);
    // 9. BN3 + residual + ReLU
    channel_stats<<<dim3(CC), dim3(256), 0, stream>>>(y3, mean3, istd3, PP);
    bn_add_relu<<<dim3((unsigned)(SZ_IMG / 256)), dim3(256), 0, stream>>>(
        y3, x, g3, b3, mean3, istd3, out);
}
